// TrainedQuantumTransformer_17583596109981
// MI455X (gfx1250) — matrix operand order repaired and checked by reading, not compiled, not run on hardware
//
#include <hip/hip_runtime.h>
#include <hip/hip_bf16.h>
#include <math.h>

// ---------------------------------------------------------------------------
// MI455X / gfx1250 TrainedQuantumTransformer forward.
// All big GEMMs on v_wmma_f32_16x16x32_f16 (f32 accumulate).
// Every f16 operand is converted into a zero-padded layout (K -> mult of 32,
// N -> mult of 64, head dim 41 -> 48) so the GEMM/attention hot loops have
// no bounds checks and all global traffic is 16/32-byte vectorized.
// ---------------------------------------------------------------------------

typedef _Float16 half_t;
typedef __attribute__((ext_vector_type(16))) _Float16 v16h;
typedef __attribute__((ext_vector_type(8)))  _Float16 v8h;
typedef __attribute__((ext_vector_type(8)))  float    v8f;

#define D_MODEL 288
#define DM_P    320          // d_model padded to mult of 64
#define NHEADS  8
#define QKVD    328
#define D_HEAD  41
#define DH_P    48           // head dim padded
#define QKV_P   (NHEADS*DH_P)   // 384
#define D_FF    1315
#define DFF_P   1344         // mult of 64
#define IN_DIM  83
#define IN_P    96           // mult of 32
#define NBLOCKS 4
#define BSZ     1024
#define SEQ     128
#define MTOK    (BSZ*SEQ)
#define EPSV    1e-5f

__device__ __forceinline__ float gelu_f(float x) {
    return 0.5f * x * (1.0f + erff(x * 0.70710678118654752f));
}

// ------------------- padded f32 -> f16 conversion kernels ------------------
// dst[L][Rp][Kp] = src[L][R][K] zero-extended.
__global__ void conv_pad_kernel(const float* __restrict__ src, half_t* __restrict__ dst,
                                int L, int R, int K, int Rp, int Kp) {
    size_t i = (size_t)blockIdx.x * blockDim.x + threadIdx.x;
    size_t total = (size_t)L * Rp * Kp;
    if (i >= total) return;
    int kp = (int)(i % Kp);
    size_t t = i / Kp;
    int rp = (int)(t % Rp);
    int l  = (int)(t / Rp);
    half_t v = (half_t)0.0f;
    if (rp < R && kp < K) v = (half_t)src[((size_t)l * R + rp) * K + kp];
    dst[i] = v;
}

// Wq/Wk/Wv [4][328][288] -> [4][384][288], rows remapped h*48+d <- h*41+d.
__global__ void conv_qkvw_kernel(const float* __restrict__ src, half_t* __restrict__ dst) {
    size_t i = (size_t)blockIdx.x * blockDim.x + threadIdx.x;
    if (i >= (size_t)NBLOCKS * QKV_P * D_MODEL) return;
    int k = (int)(i % D_MODEL);
    size_t t = i / D_MODEL;
    int np = (int)(t % QKV_P);
    int l  = (int)(t / QKV_P);
    int h = np / DH_P, d = np % DH_P;
    half_t v = (half_t)0.0f;
    if (d < D_HEAD) v = (half_t)src[((size_t)l * QKVD + h * D_HEAD + d) * D_MODEL + k];
    dst[i] = v;
}

// Wo [4][288][328] -> [4][320][384], cols remapped h*48+d <- h*41+d.
__global__ void conv_wo_kernel(const float* __restrict__ src, half_t* __restrict__ dst) {
    size_t i = (size_t)blockIdx.x * blockDim.x + threadIdx.x;
    if (i >= (size_t)NBLOCKS * DM_P * QKV_P) return;
    int kp = (int)(i % QKV_P);
    size_t t = i / QKV_P;
    int np = (int)(t % DM_P);
    int l  = (int)(t / DM_P);
    int h = kp / DH_P, d = kp % DH_P;
    half_t v = (half_t)0.0f;
    if (np < D_MODEL && d < D_HEAD)
        v = (half_t)src[((size_t)l * D_MODEL + np) * QKVD + h * D_HEAD + d];
    dst[i] = v;
}

// bias pad: dst[L][Np] = src[L][N] zero-extended.
__global__ void pad_bias_kernel(const float* __restrict__ src, float* __restrict__ dst,
                                int L, int N, int Np) {
    int i = blockIdx.x * blockDim.x + threadIdx.x;
    if (i >= L * Np) return;
    int j = i % Np, l = i / Np;
    dst[i] = (j < N) ? src[l * N + j] : 0.0f;
}

// bq/bk/bv [4][328] -> [4][384] head-remapped.
__global__ void pad_bias_qkv_kernel(const float* __restrict__ src, float* __restrict__ dst) {
    int i = blockIdx.x * blockDim.x + threadIdx.x;
    if (i >= NBLOCKS * QKV_P) return;
    int j = i % QKV_P, l = i / QKV_P;
    int h = j / DH_P, d = j % DH_P;
    dst[i] = (d < D_HEAD) ? src[l * QKVD + h * D_HEAD + d] : 0.0f;
}

// ------------------------ positional-encoding add --------------------------
__global__ void add_pe_kernel(float* __restrict__ h, const float* __restrict__ phase) {
    size_t i = (size_t)blockIdx.x * blockDim.x + threadIdx.x;
    if (i >= (size_t)MTOK * D_MODEL) return;
    int d = (int)(i % D_MODEL);
    size_t t = i / D_MODEL;
    int s = (int)(t % SEQ);
    int i2 = d >> 1;
    float div = __expf(-(float)(2 * i2) * (logf(10000.0f) / (float)D_MODEL));
    float ang = (float)s * div;
    float pe = (d & 1) ? __cosf(ang) : __sinf(ang);
    h[t * DM_P + d] += pe * __cosf(phase[d]);
}

// ------------------------------ tiled GEMM ---------------------------------
// C[M,Np] = act( A[M,Kp] * W[Np,Kp]^T + bias[Np] ).  All dims padded:
// M % 128 == 0, Np % 64 == 0, Kp % 32 == 0 -> NO bounds checks anywhere.
// 256 threads = 8 waves (4 M x 2 N), wave tile 32x32 = 2x2 WMMA accums.
// Register double-buffer: next tile loads overlap current-tile WMMAs.
#define BM 128
#define BN 64
#define BK 32

__global__ __launch_bounds__(256)
void gemm_f16_kernel(const half_t* __restrict__ A, int lda,
                     const half_t* __restrict__ W, int ldw,
                     const float*  __restrict__ bias,
                     float*  __restrict__ Cf,
                     half_t* __restrict__ Ch, int ldc,
                     int Kp, int act)
{
    __shared__ __align__(128) half_t sA[BM * BK];
    __shared__ __align__(128) half_t sB[BN * BK];

    const int tid  = threadIdx.x;
    const int lane = tid & 31;
    const int wave = tid >> 5;
    const int half_id = lane >> 4;
    const int m16     = lane & 15;
    const int wm = wave & 3;
    const int wn = wave >> 2;
    const long blockM = (long)blockIdx.y * BM;
    const long blockN = (long)blockIdx.x * BN;

    v8f acc[2][2] = {};

    const int a_r = tid >> 1, a_c0 = (tid & 1) * 16;  // 32B vector per thread
    const int b_r = tid >> 2, b_c0 = (tid & 3) * 8;   // 16B vector per thread

    const half_t* aptr = A + (blockM + a_r) * (long)lda + a_c0;
    const half_t* wptr = W + (blockN + b_r) * (long)ldw + b_c0;

    v16h a_reg = *(const v16h*)(aptr);
    v8h  b_reg = *(const v8h*)(wptr);

    for (int k0 = 0; k0 < Kp; k0 += BK) {
        *(v16h*)&sA[a_r * BK + a_c0] = a_reg;
        *(v8h*)&sB[b_r * BK + b_c0] = b_reg;
        __syncthreads();

        if (k0 + BK < Kp) {                      // prefetch next tile into regs
            a_reg = *(const v16h*)(aptr + k0 + BK);
            b_reg = *(const v8h*)(wptr + k0 + BK);
        }

        v16h af[2], bf[2];
#pragma unroll
        for (int i = 0; i < 2; ++i) {            // A frag: 16 contiguous K
            const half_t* r = &sA[(wm * 32 + i * 16 + m16) * BK + half_id * 16];
            v8h lo = *(const v8h*)r;
            v8h hi = *(const v8h*)(r + 8);
#pragma unroll
            for (int j = 0; j < 8; ++j) { af[i][j] = lo[j]; af[i][j + 8] = hi[j]; }
        }
#pragma unroll
        for (int j = 0; j < 2; ++j)              // B frag: 16 contiguous K
            bf[j] = *(const v16h*)&sB[(wn * 32 + j * 16 + m16) * BK + half_id * 16];

#pragma unroll
        for (int i = 0; i < 2; ++i)
#pragma unroll
            for (int j = 0; j < 2; ++j)
                acc[i][j] = __builtin_amdgcn_wmma_f32_16x16x32_f16(
                    false, af[i], false, bf[j], (short)0, acc[i][j], false, false);
        __syncthreads();
    }

#pragma unroll
    for (int i = 0; i < 2; ++i)
#pragma unroll
        for (int j = 0; j < 2; ++j)
#pragma unroll
            for (int r = 0; r < 8; ++r) {
                long gm = blockM + wm * 32 + i * 16 + half_id * 8 + r;
                long gn = blockN + wn * 32 + j * 16 + m16;
                float v = acc[i][j][r] + bias[gn];   // bias is padded -> no guard
                if (act == 1) v = gelu_f(v);
                if (Cf) Cf[gm * ldc + gn] = v;
                if (Ch) Ch[gm * ldc + gn] = (half_t)v;
            }
}

// --------------------------- fused attention -------------------------------
// One workgroup per (batch, head). q/k/v/o use the head-padded [tok, 8*48]
// f16 layout, so all global traffic is aligned 16B vectors. Pad columns of
// q/k are zero (padded weights+bias) so they add 0 to scores; vT pad rows
// are zero so O pad columns come out 0 (which is exactly what the Wo GEMM
// expects for its padded K).
__global__ __launch_bounds__(256)
void attn_kernel(const half_t* __restrict__ q, const half_t* __restrict__ k,
                 const half_t* __restrict__ v, half_t* __restrict__ o,
                 const float* __restrict__ temp, const float* __restrict__ qscale,
                 int l)
{
    __shared__ __align__(128) half_t smem[24576];  // 48 KB
    half_t* qs = smem;           // phase 1: [128][64]
    half_t* ks = smem + 8192;    // phase 1: [128][64]
    half_t* vT = smem + 16384;   // [64][128] (disjoint region)
    half_t* ps = smem;           // phase 2: [128][128] probabilities
    half_t* os = smem;           // phase 3: [128][48] output staging

    const int b  = blockIdx.x >> 3;
    const int hd = blockIdx.x & 7;
    const int tid = threadIdx.x;
    const int lane = tid & 31, wave = tid >> 5;
    const int half_id = lane >> 4, m16 = lane & 15;

    const size_t base = ((size_t)b * SEQ) * QKV_P + (size_t)hd * DH_P;
    const v8h vzero = {};

    // stage q, k: 128 rows x 8 v8h chunks (6 real + 2 zero) = 1024 chunks
#pragma unroll
    for (int it = 0; it < 4; ++it) {
        int cidx = tid + it * 256;
        int s = cidx >> 3, c = cidx & 7;
        v8h qv = vzero, kv = vzero;
        if (c < 6) {
            qv = *(const v8h*)(q + base + (size_t)s * QKV_P + c * 8);
            kv = *(const v8h*)(k + base + (size_t)s * QKV_P + c * 8);
        }
        *(v8h*)&qs[s * 64 + c * 8] = qv;
        *(v8h*)&ks[s * 64 + c * 8] = kv;
    }
    // stage vT[d][s]: 128 rows x 6 chunks, transposed scatter
#pragma unroll
    for (int it = 0; it < 3; ++it) {
        int cidx = tid + it * 256;
        int s = cidx / 6, c = cidx % 6;
        v8h vv = *(const v8h*)(v + base + (size_t)s * QKV_P + c * 8);
#pragma unroll
        for (int j = 0; j < 8; ++j) vT[(c * 8 + j) * 128 + s] = vv[j];
    }
    {   // zero-fill vT rows 48..63: 16 rows x 16 v8h chunks = 256
        int d = DH_P + (tid >> 4), c = tid & 15;
        *(v8h*)&vT[d * 128 + c * 8] = vzero;
    }
    __syncthreads();

    // phase 1: scores -- wave owns 16-row strip, 8 N-tiles
    v8f sc[8] = {};
#pragma unroll
    for (int k0 = 0; k0 < 64; k0 += 32) {
        v16h af;
        {
            const half_t* r = &qs[(wave * 16 + m16) * 64 + k0 + half_id * 16];
            v8h lo = *(const v8h*)r; v8h hi = *(const v8h*)(r + 8);
#pragma unroll
            for (int j = 0; j < 8; ++j) { af[j] = lo[j]; af[j + 8] = hi[j]; }
        }
#pragma unroll
        for (int t = 0; t < 8; ++t) {
            v16h bf = *(const v16h*)&ks[(t * 16 + m16) * 64 + k0 + half_id * 16];
            sc[t] = __builtin_amdgcn_wmma_f32_16x16x32_f16(
                false, af, false, bf, (short)0, sc[t], false, false);
        }
    }

    const float inv = 1.0f / (sqrtf((float)D_HEAD) * temp[l]);
    const float qsc = qscale[l * NHEADS + hd];

    // softmax: row m = wave*16 + half*8 + r -> reduce across the 16 lanes of
    // this half with xor masks 1,2,4,8 (C-matrix lane layout).
#pragma unroll
    for (int r = 0; r < 8; ++r) {
        float mx = -1e30f;
#pragma unroll
        for (int t = 0; t < 8; ++t) { sc[t][r] *= inv; mx = fmaxf(mx, sc[t][r]); }
#pragma unroll
        for (int m = 1; m <= 8; m <<= 1) mx = fmaxf(mx, __shfl_xor(mx, m, 32));
        float sum = 0.0f;
#pragma unroll
        for (int t = 0; t < 8; ++t) { float e = __expf(sc[t][r] - mx); sc[t][r] = e; sum += e; }
#pragma unroll
        for (int m = 1; m <= 8; m <<= 1) sum += __shfl_xor(sum, m, 32);
        float f = qsc / sum;
#pragma unroll
        for (int t = 0; t < 8; ++t) sc[t][r] *= f;
    }
    __syncthreads();  // all reads of qs/ks done before ps overwrites them

#pragma unroll
    for (int r = 0; r < 8; ++r)
#pragma unroll
        for (int t = 0; t < 8; ++t)
            ps[(wave * 16 + half_id * 8 + r) * 128 + t * 16 + m16] = (half_t)sc[t][r];
    __syncthreads();

    // phase 2: O = P @ V, N = 48 (3 tiles), K = 128 (4 steps)
    v8f oc[3] = {};
#pragma unroll
    for (int k0 = 0; k0 < 128; k0 += 32) {
        v16h af;
        {
            const half_t* r = &ps[(wave * 16 + m16) * 128 + k0 + half_id * 16];
            v8h lo = *(const v8h*)r; v8h hi = *(const v8h*)(r + 8);
#pragma unroll
            for (int j = 0; j < 8; ++j) { af[j] = lo[j]; af[j + 8] = hi[j]; }
        }
#pragma unroll
        for (int j = 0; j < 3; ++j) {
            v16h bf = *(const v16h*)&vT[(j * 16 + m16) * 128 + k0 + half_id * 16];
            oc[j] = __builtin_amdgcn_wmma_f32_16x16x32_f16(
                false, af, false, bf, (short)0, oc[j], false, false);
        }
    }
    __syncthreads();  // phase 2 reads of ps complete; reuse as os[128][48]
#pragma unroll
    for (int j = 0; j < 3; ++j)
#pragma unroll
        for (int r = 0; r < 8; ++r)
            os[(wave * 16 + half_id * 8 + r) * DH_P + j * 16 + m16] = (half_t)oc[j][r];
    __syncthreads();
    // vectorized global store: 128 rows x 6 chunks
#pragma unroll
    for (int it = 0; it < 3; ++it) {
        int cidx = tid + it * 256;
        int s = cidx / 6, c = cidx % 6;
        *(v8h*)(o + base + (size_t)s * QKV_P + c * 8) = *(const v8h*)&os[s * DH_P + c * 8];
    }
}

// ---------------------- residual + LayerNorm (wave/token) ------------------
// h = LN(h + s*y) * w + b ; strides DM_P; pad columns stay zero.
__global__ __launch_bounds__(256)
void resid_ln_kernel(float* __restrict__ h32, half_t* __restrict__ h16,
                     const float* __restrict__ y32,
                     const float* __restrict__ skipv, int sidx,
                     const float* __restrict__ w, const float* __restrict__ bb)
{
    const int lane = threadIdx.x & 31;
    const int wave = threadIdx.x >> 5;
    const long t = (long)blockIdx.x * 8 + wave;
    const float s = skipv ? skipv[sidx] : 1.0f;

    float vals[9];
    float sum = 0.0f;
#pragma unroll
    for (int i = 0; i < 9; ++i) {
        int d = lane + i * 32;
        float v = h32[t * DM_P + d] + s * y32[t * DM_P + d];
        vals[i] = v; sum += v;
    }
#pragma unroll
    for (int m = 1; m <= 16; m <<= 1) sum += __shfl_xor(sum, m, 32);
    float mu = sum * (1.0f / (float)D_MODEL);
    float var = 0.0f;
#pragma unroll
    for (int i = 0; i < 9; ++i) { float d0 = vals[i] - mu; var += d0 * d0; }
#pragma unroll
    for (int m = 1; m <= 16; m <<= 1) var += __shfl_xor(var, m, 32);
    float rs = rsqrtf(var * (1.0f / (float)D_MODEL) + EPSV);
#pragma unroll
    for (int i = 0; i < 9; ++i) {
        int d = lane + i * 32;
        float o = (vals[i] - mu) * rs * w[d] + bb[d];
        h32[t * DM_P + d] = o;
        h16[t * DM_P + d] = (half_t)o;
    }
}

// ------------------------------- mean pool ---------------------------------
__global__ void pool_kernel(const float* __restrict__ h32, float* __restrict__ pooled) {
    int b = blockIdx.x, d = threadIdx.x;   // blockDim = 288
    float s = 0.0f;
    for (int t = 0; t < SEQ; ++t)
        s += h32[((size_t)b * SEQ + t) * DM_P + d];
    pooled[(size_t)b * D_MODEL + d] = s * (1.0f / (float)SEQ);
}

// ------------------------------ output heads -------------------------------
__global__ void head_kernel(const float* __restrict__ pooled,
                            const float* __restrict__ bn_w, const float* __restrict__ bn_b,
                            const float* __restrict__ bn_mean, const float* __restrict__ bn_var,
                            const float* __restrict__ h1w, const float* __restrict__ h1b,
                            const float* __restrict__ h2w, const float* __restrict__ h2b,
                            const float* __restrict__ h3w, const float* __restrict__ h3b,
                            const float* __restrict__ u1w, const float* __restrict__ u1b,
                            const float* __restrict__ u2w, const float* __restrict__ u2b,
                            float* __restrict__ out)
{
    __shared__ float pz[D_MODEL], z0[D_MODEL], z1[144], z2[72], uu[72];
    const int b = blockIdx.x, tid = threadIdx.x;  // blockDim = 288
    if (tid < D_MODEL) {
        float p = pooled[(size_t)b * D_MODEL + tid];
        pz[tid] = p;
        float z = (p - bn_mean[tid]) * rsqrtf(bn_var[tid] + EPSV) * bn_w[tid] + bn_b[tid];
        z0[tid] = gelu_f(z);
    }
    __syncthreads();
    if (tid < 144) {
        float a = h1b[tid];
        for (int kk = 0; kk < D_MODEL; ++kk) a += z0[kk] * h1w[tid * D_MODEL + kk];
        z1[tid] = gelu_f(a);
    }
    if (tid >= 144 && tid < 216) {
        int oo = tid - 144;
        float a = u1b[oo];
        for (int kk = 0; kk < D_MODEL; ++kk) a += pz[kk] * u1w[oo * D_MODEL + kk];
        uu[oo] = fmaxf(a, 0.0f);
    }
    __syncthreads();
    if (tid < 72) {
        float a = h2b[tid];
        for (int kk = 0; kk < 144; ++kk) a += z1[kk] * h2w[tid * 144 + kk];
        z2[tid] = gelu_f(a);
    }
    __syncthreads();
    if (tid == 0) {
        float a = h3b[0];
        for (int kk = 0; kk < 72; ++kk) a += z2[kk] * h3w[kk];
        out[b] = a;
    }
    if (tid == 1) {
        float a = u2b[0];
        for (int kk = 0; kk < 72; ++kk) a += uu[kk] * u2w[kk];
        out[BSZ + b] = (a > 20.0f) ? a : log1pf(__expf(a));  // softplus
    }
}

// ---------------------------------------------------------------------------
extern "C" void kernel_launch(void* const* d_in, const int* in_sizes, int n_in,
                              void* d_out, int out_size, void* d_ws, size_t ws_size,
                              hipStream_t stream)
{
    const float* x     = (const float*)d_in[0];
    const float* Wp    = (const float*)d_in[1];
    const float* bp    = (const float*)d_in[2];
    const float* phase = (const float*)d_in[3];
    const float* Wq    = (const float*)d_in[4];
    const float* bq    = (const float*)d_in[5];
    const float* Wk    = (const float*)d_in[6];
    const float* bk    = (const float*)d_in[7];
    const float* Wv    = (const float*)d_in[8];
    const float* bv    = (const float*)d_in[9];
    const float* Wo    = (const float*)d_in[10];
    const float* bo    = (const float*)d_in[11];
    const float* temp  = (const float*)d_in[12];
    const float* qscl  = (const float*)d_in[13];
    const float* Wf1   = (const float*)d_in[14];
    const float* bf1   = (const float*)d_in[15];
    const float* Wf2   = (const float*)d_in[16];
    const float* bf2   = (const float*)d_in[17];
    const float* n1w   = (const float*)d_in[18];
    const float* n1b   = (const float*)d_in[19];
    const float* n2w   = (const float*)d_in[20];
    const float* n2b   = (const float*)d_in[21];
    const float* skip  = (const float*)d_in[22];
    const float* bn_w  = (const float*)d_in[23];
    const float* bn_b  = (const float*)d_in[24];
    const float* bn_m  = (const float*)d_in[25];
    const float* bn_v  = (const float*)d_in[26];
    const float* h1w   = (const float*)d_in[27];
    const float* h1b   = (const float*)d_in[28];
    const float* h2w   = (const float*)d_in[29];
    const float* h2b   = (const float*)d_in[30];
    const float* h3w   = (const float*)d_in[31];
    const float* h3b   = (const float*)d_in[32];
    const float* u1w   = (const float*)d_in[33];
    const float* u1b   = (const float*)d_in[34];
    const float* u2w   = (const float*)d_in[35];
    const float* u2b   = (const float*)d_in[36];
    float* out = (float*)d_out;
    (void)in_sizes; (void)n_in; (void)out_size; (void)ws_size;

    // ---- workspace layout (all regions fully rewritten every call) ----
    char* base = (char*)d_ws;
    size_t off = 0;
    auto alloc = [&](size_t bytes) -> void* {
        off = (off + 255) & ~(size_t)255;
        void* p = base + off;
        off += bytes;
        return p;
    };
    float*  h32  = (float*) alloc((size_t)MTOK * DM_P * 4);
    float*  y32  = (float*) alloc((size_t)MTOK * DM_P * 4);
    half_t* h16  = (half_t*)alloc((size_t)MTOK * DM_P * 2);
    half_t* x16  = (half_t*)alloc((size_t)MTOK * IN_P * 2);
    // q/k/v/o (4 * M*384*2) alias the FFN buffer (M*1344*2, smaller)
    size_t qkvsz = (size_t)MTOK * QKV_P * 2;
    char*  big   = (char*)alloc(4 * qkvsz);
    half_t* q16  = (half_t*)(big);
    half_t* k16  = (half_t*)(big + qkvsz);
    half_t* v16  = (half_t*)(big + 2 * qkvsz);
    half_t* o16  = (half_t*)(big + 3 * qkvsz);
    half_t* f16b = (half_t*)(big);
    half_t* Wp16 = (half_t*)alloc((size_t)DM_P * IN_P * 2);
    half_t* Wq16 = (half_t*)alloc((size_t)NBLOCKS * QKV_P * D_MODEL * 2);
    half_t* Wk16 = (half_t*)alloc((size_t)NBLOCKS * QKV_P * D_MODEL * 2);
    half_t* Wv16 = (half_t*)alloc((size_t)NBLOCKS * QKV_P * D_MODEL * 2);
    half_t* Wo16 = (half_t*)alloc((size_t)NBLOCKS * DM_P * QKV_P * 2);
    half_t* Wf116= (half_t*)alloc((size_t)NBLOCKS * DFF_P * D_MODEL * 2);
    half_t* Wf216= (half_t*)alloc((size_t)NBLOCKS * DM_P * DFF_P * 2);
    float*  bp_p = (float*) alloc(DM_P * 4);
    float*  bq_p = (float*) alloc((size_t)NBLOCKS * QKV_P * 4);
    float*  bk_p = (float*) alloc((size_t)NBLOCKS * QKV_P * 4);
    float*  bv_p = (float*) alloc((size_t)NBLOCKS * QKV_P * 4);
    float*  bo_p = (float*) alloc((size_t)NBLOCKS * DM_P * 4);
    float*  bf1_p= (float*) alloc((size_t)NBLOCKS * DFF_P * 4);
    float*  bf2_p= (float*) alloc((size_t)NBLOCKS * DM_P * 4);
    float*  pooled=(float*) alloc((size_t)BSZ * D_MODEL * 4);

    auto cpad = [&](const float* s, half_t* d, int L, int R, int K, int Rp, int Kp) {
        size_t n = (size_t)L * Rp * Kp;
        conv_pad_kernel<<<(unsigned)((n + 255) / 256), 256, 0, stream>>>(s, d, L, R, K, Rp, Kp);
    };
    auto bpad = [&](const float* s, float* d, int L, int N, int Np) {
        int n = L * Np;
        pad_bias_kernel<<<(n + 255) / 256, 256, 0, stream>>>(s, d, L, N, Np);
    };
    auto gemm = [&](const half_t* A, int lda, const half_t* W, int ldw,
                    const float* bias, float* Cf, half_t* Ch, int ldc,
                    int Np, int Kp, int act) {
        dim3 grid(Np / BN, MTOK / BM);
        gemm_f16_kernel<<<grid, 256, 0, stream>>>(A, lda, W, ldw, bias, Cf, Ch, ldc, Kp, act);
    };

    // ---- padded conversions (per-launch, no cross-call caching) ----
    cpad(x,   x16,  1, MTOK, IN_DIM, MTOK, IN_P);
    cpad(Wp,  Wp16, 1, D_MODEL, IN_DIM, DM_P, IN_P);
    {
        size_t n = (size_t)NBLOCKS * QKV_P * D_MODEL;
        conv_qkvw_kernel<<<(unsigned)((n + 255) / 256), 256, 0, stream>>>(Wq, Wq16);
        conv_qkvw_kernel<<<(unsigned)((n + 255) / 256), 256, 0, stream>>>(Wk, Wk16);
        conv_qkvw_kernel<<<(unsigned)((n + 255) / 256), 256, 0, stream>>>(Wv, Wv16);
        size_t m = (size_t)NBLOCKS * DM_P * QKV_P;
        conv_wo_kernel<<<(unsigned)((m + 255) / 256), 256, 0, stream>>>(Wo, Wo16);
    }
    cpad(Wf1, Wf116, NBLOCKS, D_FF, D_MODEL, DFF_P, D_MODEL);
    cpad(Wf2, Wf216, NBLOCKS, D_MODEL, D_FF, DM_P, DFF_P);
    bpad(bp,  bp_p,  1, D_MODEL, DM_P);
    pad_bias_qkv_kernel<<<(NBLOCKS * QKV_P + 255) / 256, 256, 0, stream>>>(bq, bq_p);
    pad_bias_qkv_kernel<<<(NBLOCKS * QKV_P + 255) / 256, 256, 0, stream>>>(bk, bk_p);
    pad_bias_qkv_kernel<<<(NBLOCKS * QKV_P + 255) / 256, 256, 0, stream>>>(bv, bv_p);
    bpad(bo,  bo_p,  NBLOCKS, D_MODEL, DM_P);
    bpad(bf1, bf1_p, NBLOCKS, D_FF, DFF_P);
    bpad(bf2, bf2_p, NBLOCKS, D_MODEL, DM_P);

    // ---- input projection + positional encoding ----
    gemm(x16, IN_P, Wp16, IN_P, bp_p, h32, nullptr, DM_P, DM_P, IN_P, 0);
    add_pe_kernel<<<(unsigned)(((size_t)MTOK * D_MODEL + 255) / 256), 256, 0, stream>>>(h32, phase);
    cpad(h32, h16, 1, MTOK, DM_P, MTOK, DM_P);   // straight copy (incl. zero pad)

    for (int l = 0; l < NBLOCKS; ++l) {
        gemm(h16, DM_P, Wq16 + (size_t)l * QKV_P * D_MODEL, D_MODEL,
             bq_p + l * QKV_P, nullptr, q16, QKV_P, QKV_P, D_MODEL, 0);
        gemm(h16, DM_P, Wk16 + (size_t)l * QKV_P * D_MODEL, D_MODEL,
             bk_p + l * QKV_P, nullptr, k16, QKV_P, QKV_P, D_MODEL, 0);
        gemm(h16, DM_P, Wv16 + (size_t)l * QKV_P * D_MODEL, D_MODEL,
             bv_p + l * QKV_P, nullptr, v16, QKV_P, QKV_P, D_MODEL, 0);
        attn_kernel<<<BSZ * NHEADS, 256, 0, stream>>>(q16, k16, v16, o16, temp, qscl, l);
        gemm(o16, QKV_P, Wo16 + (size_t)l * DM_P * QKV_P, QKV_P,
             bo_p + l * DM_P, y32, nullptr, DM_P, DM_P, QKV_P, 0);
        resid_ln_kernel<<<MTOK / 8, 256, 0, stream>>>(h32, h16, y32, skip, l,
                                                      n1w + l * D_MODEL, n1b + l * D_MODEL);
        gemm(h16, DM_P, Wf116 + (size_t)l * DFF_P * D_MODEL, D_MODEL,
             bf1_p + l * DFF_P, nullptr, f16b, DFF_P, DFF_P, D_MODEL, 1);
        gemm(f16b, DFF_P, Wf216 + (size_t)l * DM_P * DFF_P, DFF_P,
             bf2_p + l * DM_P, y32, nullptr, DM_P, DM_P, DFF_P, 0);
        resid_ln_kernel<<<MTOK / 8, 256, 0, stream>>>(h32, h16, y32, nullptr, 0,
                                                      n2w + l * D_MODEL, n2b + l * D_MODEL);
    }

    pool_kernel<<<BSZ, D_MODEL, 0, stream>>>(h32, pooled);
    head_kernel<<<BSZ, D_MODEL, 0, stream>>>(pooled, bn_w, bn_b, bn_m, bn_v,
                                             h1w, h1b, h2w, h2b, h3w, h3b,
                                             u1w, u1b, u2w, u2b, out);
}
